// GCATopo_37993280700931
// MI455X (gfx1250) — matrix operand order repaired and checked by reading, not compiled, not run on hardware
//
#include <hip/hip_runtime.h>

// ---------------------------------------------------------------------------
// GTAT GNN forward for MI455X (gfx1250, wave32).
// Dense GEMMs (~50 GFLOP) run through v_wmma_f32_16x16x32_bf16 (f32 accum).
// Edge softmax/scatter phase is L2-resident atomics (all tables << 192MB L2).
// ---------------------------------------------------------------------------

#define NEG_SLOPE 0.2f

typedef __bf16 bf16_t;
typedef __attribute__((ext_vector_type(8)))  __bf16 v8bf;
typedef __attribute__((ext_vector_type(16))) __bf16 v16bf;
typedef __attribute__((ext_vector_type(8)))  float  v8f;

__device__ __forceinline__ unsigned short f2bf(float f) {
  unsigned u = __float_as_uint(f);
  u += 0x7FFFu + ((u >> 16) & 1u);           // round-to-nearest-even
  return (unsigned short)(u >> 16);
}
__device__ __forceinline__ unsigned ord_enc(float f) {  // monotone float->uint
  unsigned u = __float_as_uint(f);
  return (u & 0x80000000u) ? ~u : (u | 0x80000000u);
}
__device__ __forceinline__ float ord_dec(unsigned u) {
  return (u & 0x80000000u) ? __uint_as_float(u ^ 0x80000000u)
                           : __uint_as_float(~u);
}
__device__ __forceinline__ float leaky(float x) { return x > 0.f ? x : NEG_SLOPE * x; }

// -------------------------------- utility kernels --------------------------

__global__ void k_zero(unsigned* __restrict__ p, long long n) {
  long long i = (long long)blockIdx.x * blockDim.x + threadIdx.x;
  if (i < n) p[i] = 0u;
}

__global__ void k_f2bf(const float* __restrict__ in, unsigned short* __restrict__ out,
                       long long n) {
  long long i = (long long)blockIdx.x * blockDim.x + threadIdx.x;
  if (i < n) out[i] = f2bf(in[i]);
}

// W[K,N] fp32  ->  Wt[N,K] bf16  (so B fragments are contiguous per lane)
__global__ void k_transpose_bf(const float* __restrict__ W, unsigned short* __restrict__ Wt,
                               int K, int N) {
  long long i = (long long)blockIdx.x * blockDim.x + threadIdx.x;
  if (i >= (long long)K * N) return;
  int k = (int)(i / N);
  int n = (int)(i - (long long)k * N);
  Wt[(size_t)n * K + k] = f2bf(W[i]);
}

// -------------------------------- WMMA GEMM --------------------------------
// C[M,N] = A[M,K](bf16) @ B[K,N] + bias, B given transposed as Bt[N,K](bf16).
// One wave -> 16x64 output strip: 1 A fragment, 4 B fragments, 4 WMMA / K-step.
// Requires: K % 32 == 0, N % 64 == 0, M % 16 == 0 (true for all call sites).

__device__ __forceinline__ v16bf ld_fragA(const bf16_t* __restrict__ p) {
  // ISA 16-bit A layout: lane half 'kh' holds K = kh*8..+7 and K = 16+kh*8..+7
  v8bf lo = *(const v8bf*)(p);
  v8bf hi = *(const v8bf*)(p + 16);
  v16bf r;
#pragma unroll
  for (int i = 0; i < 8; ++i) { r[i] = lo[i]; r[i + 8] = hi[i]; }
  return r;
}

__global__ void k_gemm_bf16(const unsigned short* __restrict__ Abf,
                            const unsigned short* __restrict__ Btbf,
                            const float* __restrict__ bias,
                            float* __restrict__ C,
                            int M, int N, int K, int relu) {
  const int wave  = (int)((blockIdx.x * blockDim.x + threadIdx.x) >> 5);
  const int lane  = threadIdx.x & 31;
  const int strips = N >> 6;
  const int mrow  = wave / strips;
  const int strip = wave - mrow * strips;
  if (mrow * 16 >= M) return;                 // wave-uniform: EXEC stays all-ones
  const int m0 = mrow << 4;
  const int n0 = strip << 6;
  const int ml = lane & 15;
  const int kh = lane >> 4;

  const bf16_t* A  = (const bf16_t*)Abf  + (size_t)(m0 + ml) * K + kh * 16;
  const bf16_t* B0 = (const bf16_t*)Btbf + (size_t)(n0 + ml) * K + kh * 16;
  const bf16_t* B1 = B0 + (size_t)16 * K;
  const bf16_t* B2 = B0 + (size_t)32 * K;
  const bf16_t* B3 = B0 + (size_t)48 * K;

  v8f c0 = {0.f,0.f,0.f,0.f,0.f,0.f,0.f,0.f};
  v8f c1 = c0, c2 = c0, c3 = c0;

  for (int kb = 0; kb < K; kb += 32) {
    v16bf a  = *(const v16bf*)(A + kb);
    // B layout: lane half holds contiguous K = kh*16..+15 -> one 32B load
    v16bf b0 = *(const v16bf*)(B0 + kb);
    v16bf b1 = *(const v16bf*)(B1 + kb);
    v16bf b2 = *(const v16bf*)(B2 + kb);
    v16bf b3 = *(const v16bf*)(B3 + kb);
    c0 = __builtin_amdgcn_wmma_f32_16x16x32_bf16(false, a, false, b0, (short)0, c0, false, false);
    c1 = __builtin_amdgcn_wmma_f32_16x16x32_bf16(false, a, false, b1, (short)0, c1, false, false);
    c2 = __builtin_amdgcn_wmma_f32_16x16x32_bf16(false, a, false, b2, (short)0, c2, false, false);
    c3 = __builtin_amdgcn_wmma_f32_16x16x32_bf16(false, a, false, b3, (short)0, c3, false, false);
  }

  const int col = n0 + ml;
  const float bv0 = bias[col], bv1 = bias[col + 16], bv2 = bias[col + 32], bv3 = bias[col + 48];
#pragma unroll
  for (int r = 0; r < 8; ++r) {
    size_t base = (size_t)(m0 + r + kh * 8) * N;
    float v0 = c0[r] + bv0, v1 = c1[r] + bv1, v2 = c2[r] + bv2, v3 = c3[r] + bv3;
    if (relu) { v0 = fmaxf(v0, 0.f); v1 = fmaxf(v1, 0.f); v2 = fmaxf(v2, 0.f); v3 = fmaxf(v3, 0.f); }
    C[base + col]      = v0;
    C[base + col + 16] = v1;
    C[base + col + 32] = v2;
    C[base + col + 48] = v3;
  }
}

// -------------------------------- small dense ops --------------------------

// topo = hid[N,128] @ W[128,15] + b   (negligible FLOPs -> plain VALU)
__global__ void k_topo_proj(const float* __restrict__ hid, const float* __restrict__ W,
                            const float* __restrict__ b, float* __restrict__ out, int N) {
  int i = blockIdx.x * blockDim.x + threadIdx.x;
  if (i >= N * 15) return;
  int n = i / 15, t = i - n * 15;
  const float* hr = hid + (size_t)n * 128;
  float acc = b[t];
#pragma unroll 8
  for (int k = 0; k < 128; ++k) acc = fmaf(hr[k], W[k * 15 + t], acc);
  out[i] = acc;
}

// -------------------------------- edge phase -------------------------------

__global__ void k_edge_logits(const int* __restrict__ ei, int E, int N,
                              const float* __restrict__ xl, const float* __restrict__ xr,
                              const float* __restrict__ topo,
                              const float* __restrict__ att, const float* __restrict__ att2,
                              float* __restrict__ a1, float* __restrict__ a2,
                              unsigned* __restrict__ m1, unsigned* __restrict__ m2) {
  int i = blockIdx.x * blockDim.x + threadIdx.x;
  int EE = E + N;
  if (i >= EE * 4) return;
  int e = i >> 2, h = i & 3;
  int s = (e < E) ? ei[e]     : e - E;
  int d = (e < E) ? ei[E + e] : e - E;
  const float* xs = xl + (size_t)s * 512 + h * 128;
  const float* xd = xr + (size_t)d * 512 + h * 128;
  const float* at = att + h * 128;
  float acc = 0.f;
#pragma unroll 8
  for (int c = 0; c < 128; ++c) acc = fmaf(xs[c] + xd[c], at[c], acc);
  float v1 = leaky(acc);
  const float* ts = topo + (size_t)s * 15;
  const float* td = topo + (size_t)d * 15;
  const float* at2 = att2 + h * 15;
  float acc2 = 0.f;
#pragma unroll
  for (int t = 0; t < 15; ++t) acc2 = fmaf(ts[t] + td[t], at2[t], acc2);
  float v2 = leaky(acc2);
  a1[i] = v1; a2[i] = v2;
  atomicMax(&m1[d * 4 + h], ord_enc(v1));
  atomicMax(&m2[d * 4 + h], ord_enc(v2));
}

__global__ void k_edge_expsum(const int* __restrict__ ei, int E, int N,
                              float* __restrict__ a1, float* __restrict__ a2,
                              const unsigned* __restrict__ m1, const unsigned* __restrict__ m2,
                              float* __restrict__ s1, float* __restrict__ s2) {
  int i = blockIdx.x * blockDim.x + threadIdx.x;
  int EE = E + N;
  if (i >= EE * 4) return;
  int e = i >> 2, h = i & 3;
  int d = (e < E) ? ei[E + e] : e - E;
  float e1 = __expf(a1[i] - ord_dec(m1[d * 4 + h]));
  float e2 = __expf(a2[i] - ord_dec(m2[d * 4 + h]));
  a1[i] = e1; a2[i] = e2;
  atomicAdd(&s1[d * 4 + h], e1);
  atomicAdd(&s2[d * 4 + h], e2);
}

// Feature message weighted by TOPO attention (a2) — stream swap per reference.
__global__ void k_scatter_feat(const int* __restrict__ ei, int E, int N,
                               const float* __restrict__ xl,
                               const float* __restrict__ a2, const float* __restrict__ s2,
                               float* __restrict__ aggF) {
  long long i = (long long)blockIdx.x * blockDim.x + threadIdx.x;
  if (i >= (long long)(E + N) * 512) return;
  int e = (int)(i >> 9), c = (int)(i & 511), h = c >> 7;
  int s = (e < E) ? ei[e]     : e - E;
  int d = (e < E) ? ei[E + e] : e - E;
  float w = a2[e * 4 + h] / (s2[d * 4 + h] + 1e-16f);
  atomicAdd(&aggF[(size_t)d * 512 + c], xl[(size_t)s * 512 + c] * w);
}

// Topo message weighted by FEATURE attention (a1).
__global__ void k_scatter_topo(const int* __restrict__ ei, int E, int N,
                               const float* __restrict__ topoIn,
                               const float* __restrict__ a1, const float* __restrict__ s1,
                               float* __restrict__ aggT) {
  long long i = (long long)blockIdx.x * blockDim.x + threadIdx.x;
  if (i >= (long long)(E + N) * 60) return;
  int e = (int)(i / 60);
  int r = (int)(i - (long long)e * 60);
  int h = r / 15, t = r - h * 15;
  int s = (e < E) ? ei[e]     : e - E;
  int d = (e < E) ? ei[E + e] : e - E;
  float w = a1[e * 4 + h] / (s1[d * 4 + h] + 1e-16f);
  atomicAdd(&aggT[(size_t)d * 60 + r], topoIn[(size_t)s * 15 + t] * w);
}

__global__ void k_final_feat(const float* __restrict__ aggF, const float* __restrict__ bias,
                             float* __restrict__ out, int N) {
  long long i = (long long)blockIdx.x * blockDim.x + threadIdx.x;
  if (i >= (long long)N * 512) return;
  out[i] = aggF[i] + bias[(int)(i & 511)];
}

__global__ void k_final_topo(const float* __restrict__ aggT, const float* __restrict__ bias2,
                             float* __restrict__ out, int N) {
  int i = blockIdx.x * blockDim.x + threadIdx.x;
  if (i >= N * 15) return;
  int n = i / 15, t = i - n * 15;
  const float* p = aggT + (size_t)n * 60;
  out[i] = 0.25f * (p[t] + p[15 + t] + p[30 + t] + p[45 + t]) + bias2[t];
}

// -------------------------------- readout ----------------------------------

__global__ void k_colmean(const float* __restrict__ h, float* __restrict__ pooled,
                          int n, int cols) {
  int j = blockIdx.x;
  int t = threadIdx.x;
  float s = 0.f;
  for (int r = t; r < n; r += blockDim.x) s += h[(size_t)r * cols + j];
  __shared__ float sm[256];
  sm[t] = s;
  __syncthreads();
  for (int k = 128; k > 0; k >>= 1) { if (t < k) sm[t] += sm[t + k]; __syncthreads(); }
  if (t == 0) pooled[j] = sm[0] / (float)n;
}

// out = relu(pooled[512] @ w1[512,128] + b1) @ w2[128] + b2  (single block)
__global__ void k_head(const float* __restrict__ pooled,
                       const float* __restrict__ w1, const float* __restrict__ b1,
                       const float* __restrict__ w2, const float* __restrict__ b2,
                       float* __restrict__ out) {
  int i = threadIdx.x;   // 128 threads
  __shared__ float red[128];
  float a = b1[i];
  for (int k = 0; k < 512; ++k) a = fmaf(pooled[k], w1[k * 128 + i], a);
  red[i] = fmaxf(a, 0.f) * w2[i];
  __syncthreads();
  for (int k = 64; k > 0; k >>= 1) { if (i < k) red[i] += red[i + k]; __syncthreads(); }
  if (i == 0) out[0] = red[0] + b2[0];
}

// -------------------------------- host side --------------------------------

static inline int cdiv(long long a, int b) { return (int)((a + b - 1) / b); }

static void launch_gemm(const unsigned short* A, const unsigned short* Bt,
                        const float* bias, float* C, int M, int N, int K, int relu,
                        hipStream_t st) {
  long long waves = (long long)(M / 16) * (N / 64);
  k_gemm_bf16<<<cdiv(waves, 8), 256, 0, st>>>(A, Bt, bias, C, M, N, K, relu);
}

static void run_gtat_attention(const int* ei, int E, int N,
                               const float* xl, const float* xr, const float* topoIn,
                               const float* att, const float* att2,
                               const float* bias, const float* bias2,
                               float* a1, float* a2, unsigned* m1, unsigned* m2,
                               float* s1, float* s2, float* aggF, float* aggT,
                               float* hOut, float* topoOut, hipStream_t st) {
  const long long EE = (long long)E + N;
  k_zero<<<cdiv((long long)N * 4, 256), 256, 0, st>>>(m1, (long long)N * 4);
  k_zero<<<cdiv((long long)N * 4, 256), 256, 0, st>>>(m2, (long long)N * 4);
  k_zero<<<cdiv((long long)N * 4, 256), 256, 0, st>>>((unsigned*)s1, (long long)N * 4);
  k_zero<<<cdiv((long long)N * 4, 256), 256, 0, st>>>((unsigned*)s2, (long long)N * 4);
  k_zero<<<cdiv((long long)N * 512, 256), 256, 0, st>>>((unsigned*)aggF, (long long)N * 512);
  k_zero<<<cdiv((long long)N * 60, 256), 256, 0, st>>>((unsigned*)aggT, (long long)N * 60);

  k_edge_logits<<<cdiv(EE * 4, 256), 256, 0, st>>>(ei, E, N, xl, xr, topoIn, att, att2,
                                                   a1, a2, m1, m2);
  k_edge_expsum<<<cdiv(EE * 4, 256), 256, 0, st>>>(ei, E, N, a1, a2, m1, m2, s1, s2);
  k_scatter_feat<<<cdiv(EE * 512, 256), 256, 0, st>>>(ei, E, N, xl, a2, s2, aggF);
  k_scatter_topo<<<cdiv(EE * 60, 256), 256, 0, st>>>(ei, E, N, topoIn, a1, s1, aggT);
  k_final_feat<<<cdiv((long long)N * 512, 256), 256, 0, st>>>(aggF, bias, hOut, N);
  k_final_topo<<<cdiv((long long)N * 15, 256), 256, 0, st>>>(aggT, bias2, topoOut, N);
}

extern "C" void kernel_launch(void* const* d_in, const int* in_sizes, int n_in,
                              void* d_out, int out_size, void* d_ws, size_t ws_size,
                              hipStream_t stream) {
  const float* x      = (const float*)d_in[0];
  const int*   ei     = (const int*)d_in[1];
  const float* te_w1  = (const float*)d_in[2];
  const float* te_b1  = (const float*)d_in[3];
  const float* te_w2  = (const float*)d_in[4];
  const float* te_b2  = (const float*)d_in[5];
  const float* l1_wl  = (const float*)d_in[6];
  const float* l1_bl  = (const float*)d_in[7];
  const float* l1_wr  = (const float*)d_in[8];
  const float* l1_br  = (const float*)d_in[9];
  const float* l1_att = (const float*)d_in[10];
  const float* l1_at2 = (const float*)d_in[11];
  const float* l1_bs  = (const float*)d_in[12];
  const float* l1_bs2 = (const float*)d_in[13];
  const float* l2_wl  = (const float*)d_in[14];
  const float* l2_bl  = (const float*)d_in[15];
  const float* l2_wr  = (const float*)d_in[16];
  const float* l2_br  = (const float*)d_in[17];
  const float* l2_att = (const float*)d_in[18];
  const float* l2_at2 = (const float*)d_in[19];
  const float* l2_bs  = (const float*)d_in[20];
  const float* l2_bs2 = (const float*)d_in[21];
  const float* v_w1   = (const float*)d_in[22];
  const float* v_b1   = (const float*)d_in[23];
  const float* v_w2   = (const float*)d_in[24];
  const float* v_b2   = (const float*)d_in[25];
  const float* a_w1   = (const float*)d_in[26];
  const float* a_b1   = (const float*)d_in[27];
  const float* a_w2   = (const float*)d_in[28];
  const float* a_b2   = (const float*)d_in[29];
  float* out = (float*)d_out;

  const int N  = in_sizes[0] / 576;
  const int E  = in_sizes[1] / 2;
  const long long EE = (long long)E + N;

  // ---- carve workspace (all 256B aligned) ----
  char* w = (char*)d_ws;
  auto carve = [&](size_t bytes) -> char* {
    char* p = w; w += (bytes + 255) & ~(size_t)255; return p;
  };
  unsigned short* xbf   = (unsigned short*)carve((size_t)N * 576 * 2); // bf16 activations
  unsigned short* wT    = (unsigned short*)carve((size_t)576 * 512 * 2); // bf16 transposed W
  float* hidT  = (float*)carve((size_t)N * 128 * 4);
  float* topoA = (float*)carve((size_t)N * 15 * 4);
  float* topoB = (float*)carve((size_t)N * 15 * 4);
  float* xl    = (float*)carve((size_t)N * 512 * 4);
  float* xr    = (float*)carve((size_t)N * 512 * 4);
  float* h1    = (float*)carve((size_t)N * 512 * 4);
  float* aggF  = (float*)carve((size_t)N * 512 * 4);
  float* aggT  = (float*)carve((size_t)N * 60 * 4);
  float* a1    = (float*)carve((size_t)EE * 4 * 4);
  float* a2    = (float*)carve((size_t)EE * 4 * 4);
  unsigned* m1 = (unsigned*)carve((size_t)N * 4 * 4);
  unsigned* m2 = (unsigned*)carve((size_t)N * 4 * 4);
  float* s1    = (float*)carve((size_t)N * 4 * 4);
  float* s2    = (float*)carve((size_t)N * 4 * 4);
  float* pooled = (float*)carve(512 * 4);
  float* h2 = xl;  // safe alias: layer-2 finalize runs after scatter consumed xl

  // ---- activations -> bf16 ----
  k_f2bf<<<cdiv((long long)N * 576, 256), 256, 0, stream>>>(x, xbf, (long long)N * 576);

  // ---- topology extractor: relu(x@te_w1+b1) @ te_w2 + b2 ----
  k_transpose_bf<<<cdiv(576 * 128, 256), 256, 0, stream>>>(te_w1, wT, 576, 128);
  launch_gemm(xbf, wT, te_b1, hidT, N, 128, 576, /*relu=*/1, stream);
  k_topo_proj<<<cdiv((long long)N * 15, 256), 256, 0, stream>>>(hidT, te_w2, te_b2, topoA, N);

  // ---- layer 1 linear projections ----
  k_transpose_bf<<<cdiv(576 * 512, 256), 256, 0, stream>>>(l1_wl, wT, 576, 512);
  launch_gemm(xbf, wT, l1_bl, xl, N, 512, 576, 0, stream);
  k_transpose_bf<<<cdiv(576 * 512, 256), 256, 0, stream>>>(l1_wr, wT, 576, 512);
  launch_gemm(xbf, wT, l1_br, xr, N, 512, 576, 0, stream);

  run_gtat_attention(ei, E, N, xl, xr, topoA, l1_att, l1_at2, l1_bs, l1_bs2,
                     a1, a2, m1, m2, s1, s2, aggF, aggT, h1, topoB, stream);

  // ---- layer 2 ----
  k_f2bf<<<cdiv((long long)N * 512, 256), 256, 0, stream>>>(h1, xbf, (long long)N * 512);
  k_transpose_bf<<<cdiv(512 * 512, 256), 256, 0, stream>>>(l2_wl, wT, 512, 512);
  launch_gemm(xbf, wT, l2_bl, xl, N, 512, 512, 0, stream);
  k_transpose_bf<<<cdiv(512 * 512, 256), 256, 0, stream>>>(l2_wr, wT, 512, 512);
  launch_gemm(xbf, wT, l2_br, xr, N, 512, 512, 0, stream);

  run_gtat_attention(ei, E, N, xl, xr, topoB, l2_att, l2_at2, l2_bs, l2_bs2,
                     a1, a2, m1, m2, s1, s2, aggF, aggT, h2, topoA, stream);

  // ---- readout: mean-pool + two MLP heads ----
  k_colmean<<<512, 256, 0, stream>>>(h2, pooled, N, 512);
  k_head<<<1, 128, 0, stream>>>(pooled, v_w1, v_b1, v_w2, v_b2, out + 0);
  k_head<<<1, 128, 0, stream>>>(pooled, a_w1, a_b1, a_w2, a_b2, out + 1);
}
